// pooolingMultipleChannelAttention_74337293959581
// MI455X (gfx1250) — matrix operand order repaired and checked by reading, not compiled, not run on hardware
//
#include <hip/hip_runtime.h>
#include <hip/hip_bf16.h>

// ---------------------------------------------------------------------------
// Problem constants (from reference setup_inputs)
// ---------------------------------------------------------------------------
#define B_   2
#define C_   4
#define T_   512
#define KV_  1024
#define E_   1024
#define H_   16
#define DH_  64

typedef __attribute__((ext_vector_type(16))) _Float16 v16h;
typedef __attribute__((ext_vector_type(8)))  _Float16 v8h;
typedef __attribute__((ext_vector_type(8)))  float    v8f;

static __device__ __forceinline__ v8f wmma_f16(v16h a, v16h b, v8f c) {
  // D = A(16x32,f16) x B(32x16,f16) + C(16x16,f32)
  return __builtin_amdgcn_wmma_f32_16x16x32_f16(
      /*neg_a=*/false, a, /*neg_b=*/false, b,
      /*c_mod=*/(short)0, c, /*reuse_a=*/false, /*reuse_b=*/false);
}

static __device__ __forceinline__ void cvt4(v16h& a, int base, float4 x) {
  a[base + 0] = (_Float16)x.x; a[base + 1] = (_Float16)x.y;
  a[base + 2] = (_Float16)x.z; a[base + 3] = (_Float16)x.w;
}

static __device__ __forceinline__ v16h cat8(v8h lo, v8h hi) {
  v16h r;
#pragma unroll
  for (int i = 0; i < 8; ++i) { r[i] = lo[i]; r[8 + i] = hi[i]; }
  return r;
}

// ---------------------------------------------------------------------------
// Projection GEMM: Y = X @ W^T + bias  (X:[M,E] f32 row-major, W:[E,E] f32)
// One wave -> 16(M) x 64(N) output tile (4 WMMA accumulators, A reused 4x).
// MODE 0: Q  -> f16 q[b,h,t,d]      (scaled by Dh^-0.5)
// MODE 1: K  -> f16 k[b,h,c,k,d]    (d contiguous: score A-fragments)
// MODE 2: V  -> f16 vT[b,h,c,d,k]   (k contiguous: attend A-fragments)
// ---------------------------------------------------------------------------
template <int MODE>
__launch_bounds__(128)
__global__ void proj_kernel(const float* __restrict__ X,
                            const float* __restrict__ W,
                            const float* __restrict__ bias,
                            _Float16* __restrict__ Y) {
  const int lane  = threadIdx.x & 31;
  const int wave  = threadIdx.x >> 5;
  const int tcol  = lane & 15;   // A row / B col / D col index
  const int half  = lane >> 4;   // lane half selects K sub-range
  const int mtile = blockIdx.y * 4 + wave;
  const int nbase = blockIdx.x * 64;

  const float* xrow = X + (size_t)(mtile * 16 + tcol) * E_;

  v8f acc[4] = {};
#pragma unroll 2
  for (int k0 = 0; k0 < E_; k0 += 32) {
    // A fragment: lane half h holds K = {8h..8h+7} U {16+8h..16+8h+7}
    v16h a;
    {
      const float* p = xrow + k0 + 8 * half;
      cvt4(a, 0,  *(const float4*)(p));
      cvt4(a, 4,  *(const float4*)(p + 4));
      cvt4(a, 8,  *(const float4*)(p + 16));
      cvt4(a, 12, *(const float4*)(p + 20));
    }
#pragma unroll
    for (int s = 0; s < 4; ++s) {
      // B fragment: B(k,n) = W[n,k]; lane half h holds K = {8h..8h+7} U {16+8h..16+8h+7}
      const float* wrow = W + (size_t)(nbase + s * 16 + tcol) * E_ + k0 + 8 * half;
      v16h bf;
      cvt4(bf, 0,  *(const float4*)(wrow));
      cvt4(bf, 4,  *(const float4*)(wrow + 4));
      cvt4(bf, 8,  *(const float4*)(wrow + 16));
      cvt4(bf, 12, *(const float4*)(wrow + 20));
      acc[s] = wmma_f16(a, bf, acc[s]);
    }
  }

  const float scale = (MODE == 0) ? 0.125f : 1.0f;  // Dh^-0.5 = 64^-0.5
#pragma unroll
  for (int s = 0; s < 4; ++s) {
    const int n  = nbase + s * 16 + tcol;
    const float bv = bias[n];
    const int h = n >> 6, d = n & 63;
#pragma unroll
    for (int j = 0; j < 8; ++j) {
      const int r = mtile * 16 + j + 8 * half;  // global output row
      const float val = (acc[s][j] + bv) * scale;
      size_t idx;
      if (MODE == 0) {
        const int b = r / T_, t = r % T_;
        idx = (((size_t)b * H_ + h) * T_ + t) * DH_ + d;
      } else {
        const int b = r / (C_ * KV_);
        const int rem = r % (C_ * KV_);
        const int c = rem / KV_, k = rem % KV_;
        if (MODE == 1)
          idx = ((((size_t)b * H_ + h) * C_ + c) * KV_ + k) * DH_ + d;
        else
          idx = ((((size_t)b * H_ + h) * C_ + c) * DH_ + d) * KV_ + k;
      }
      Y[idx] = (_Float16)val;
    }
  }
}

// ---------------------------------------------------------------------------
// Output projection: out[r,n] = attn[r,:] @ Wo[n,:] + bo[n]   (f32 out)
// ---------------------------------------------------------------------------
__launch_bounds__(128)
__global__ void proj_out_kernel(const float* __restrict__ X,
                                const float* __restrict__ W,
                                const float* __restrict__ bias,
                                float* __restrict__ Y) {
  const int lane  = threadIdx.x & 31;
  const int wave  = threadIdx.x >> 5;
  const int tcol  = lane & 15;
  const int half  = lane >> 4;
  const int mtile = blockIdx.y * 4 + wave;
  const int nbase = blockIdx.x * 64;

  const float* xrow = X + (size_t)(mtile * 16 + tcol) * E_;

  v8f acc[4] = {};
#pragma unroll 2
  for (int k0 = 0; k0 < E_; k0 += 32) {
    v16h a;
    {
      const float* p = xrow + k0 + 8 * half;
      cvt4(a, 0,  *(const float4*)(p));
      cvt4(a, 4,  *(const float4*)(p + 4));
      cvt4(a, 8,  *(const float4*)(p + 16));
      cvt4(a, 12, *(const float4*)(p + 20));
    }
#pragma unroll
    for (int s = 0; s < 4; ++s) {
      const float* wrow = W + (size_t)(nbase + s * 16 + tcol) * E_ + k0 + 8 * half;
      v16h bf;
      cvt4(bf, 0,  *(const float4*)(wrow));
      cvt4(bf, 4,  *(const float4*)(wrow + 4));
      cvt4(bf, 8,  *(const float4*)(wrow + 16));
      cvt4(bf, 12, *(const float4*)(wrow + 20));
      acc[s] = wmma_f16(a, bf, acc[s]);
    }
  }
#pragma unroll
  for (int s = 0; s < 4; ++s) {
    const int n = nbase + s * 16 + tcol;
    const float bv = bias[n];
#pragma unroll
    for (int j = 0; j < 8; ++j) {
      const int r = mtile * 16 + j + 8 * half;
      Y[(size_t)r * E_ + n] = acc[s][j] + bv;
    }
  }
}

// ---------------------------------------------------------------------------
// Flash attention with channel pooling.
// One wave handles a 16-query tile for one (b,h); loops channels c, and over
// KV in 32-wide chunks with online softmax.
// Scores computed transposed: S^T = K_chunk(32xDh) @ Q^T(Dhx16), so each
// (lane, lane^16) pair holds all K-positions of one query column ->
// row-reductions need a single shfl_xor(16).
// O^T(Dh x 16) += V^T(Dh x 32) @ P^T(32 x 16).
// ---------------------------------------------------------------------------
__launch_bounds__(128)
__global__ void attn_kernel(const _Float16* __restrict__ qbf,
                            const _Float16* __restrict__ kbf,
                            const _Float16* __restrict__ vtb,
                            float* __restrict__ attnOut) {
  const int lane = threadIdx.x & 31;
  const int wave = threadIdx.x >> 5;
  const int tcol = lane & 15;
  const int half = lane >> 4;
  const int b = blockIdx.z, h = blockIdx.y;
  const int ttile = blockIdx.x * 4 + wave;
  const int t = ttile * 16 + tcol;

  // Q^T B-fragments (loop-invariant): 2 reduction steps of 32 over Dh=64.
  const _Float16* qrow = qbf + (((size_t)b * H_ + h) * T_ + t) * DH_;
  v16h qb[2];
#pragma unroll
  for (int ds = 0; ds < 2; ++ds) {
    const v8h* p = (const v8h*)(qrow + ds * 32 + 8 * half);
    qb[ds] = cat8(p[0], p[2]);
  }

  v8f tot[4] = {};

  for (int c = 0; c < C_; ++c) {
    const _Float16* kbase = kbf + ((((size_t)b * H_ + h) * C_ + c) * KV_) * DH_;
    const _Float16* vbase = vtb + ((((size_t)b * H_ + h) * C_ + c) * DH_) * KV_;

    float mrun = -1e30f, lrun = 0.0f;
    v8f o[4] = {};

    for (int kc = 0; kc < KV_; kc += 32) {
      // ---- S^T tiles: two 16(k) x 16(t) WMMAs per 32-chunk ----
      v8f st[2];
#pragma unroll
      for (int sub = 0; sub < 2; ++sub) {
        v8f s = {};
        const _Float16* krow = kbase + (size_t)(kc + sub * 16 + tcol) * DH_;
#pragma unroll
        for (int ds = 0; ds < 2; ++ds) {
          const v8h* p0 = (const v8h*)(krow + ds * 32 + 8 * half);
          const v8h* p1 = (const v8h*)(krow + ds * 32 + 16 + 8 * half);
          s = wmma_f16(cat8(*p0, *p1), qb[ds], s);
        }
        st[sub] = s;
      }

      // ---- online softmax over this 32-chunk ----
      float mloc = st[0][0];
#pragma unroll
      for (int j = 0; j < 8; ++j) {
        mloc = fmaxf(mloc, st[0][j]);
        mloc = fmaxf(mloc, st[1][j]);
      }
      mloc = fmaxf(mloc, __shfl_xor(mloc, 16, 32));
      const float mnew  = fmaxf(mrun, mloc);
      const float alpha = __expf(mrun - mnew);

      float p0a[8], p1a[8], ssum = 0.0f;
#pragma unroll
      for (int j = 0; j < 8; ++j) {
        p0a[j] = __expf(st[0][j] - mnew); ssum += p0a[j];
        p1a[j] = __expf(st[1][j] - mnew); ssum += p1a[j];
      }
      ssum += __shfl_xor(ssum, 16, 32);
      lrun = lrun * alpha + ssum;
#pragma unroll
      for (int dt = 0; dt < 4; ++dt)
#pragma unroll
        for (int j = 0; j < 8; ++j) o[dt][j] *= alpha;

      // ---- repack P^T (D-layout f32) into a B-fragment (f16) ----
      // D layout: half0 holds k {0..7}(sub0),{16..23}(sub1);
      //           half1 holds k {8..15}(sub0),{24..31}(sub1).
      // B-frag:   this code uses half0 k 0..15, half1 k 16..31 (as the V^T A-frag).
      float q0[8], q1[8];
#pragma unroll
      for (int j = 0; j < 8; ++j) {
        q0[j] = __shfl_xor(p0a[j], 16, 32);
        q1[j] = __shfl_xor(p1a[j], 16, 32);
      }
      v16h pb;
#pragma unroll
      for (int j = 0; j < 8; ++j) {
        pb[j]     = (_Float16)(half ? q1[j] : p0a[j]);
        pb[8 + j] = (_Float16)(half ? p1a[j] : q0[j]);
      }

      // ---- O^T += V^T @ P^T : 4 d-tiles over Dh=64, reduction 32 ----
#pragma unroll
      for (int dt = 0; dt < 4; ++dt) {
        const _Float16* vrow = vbase + (size_t)(dt * 16 + tcol) * KV_ + kc;
        const v8h* p0 = (const v8h*)(vrow + 16 * half);
        const v8h* p1 = (const v8h*)(vrow + 16 * half + 8);
        o[dt] = wmma_f16(cat8(*p0, *p1), pb, o[dt]);
      }
      mrun = mnew;
    }

    const float inv = 1.0f / (lrun * (float)C_);  // softmax norm * channel weight
#pragma unroll
    for (int dt = 0; dt < 4; ++dt)
#pragma unroll
      for (int j = 0; j < 8; ++j) tot[dt][j] += o[dt][j] * inv;
  }

  // Write attn output as f32 [B,T,E] for the final projection.
  float* orow = attnOut + ((size_t)b * T_ + t) * E_ + h * DH_;
#pragma unroll
  for (int dt = 0; dt < 4; ++dt)
#pragma unroll
    for (int j = 0; j < 8; ++j)
      orow[dt * 16 + j + 8 * half] = tot[dt][j];
}

// ---------------------------------------------------------------------------
// Host launch
// ---------------------------------------------------------------------------
extern "C" void kernel_launch(void* const* d_in, const int* in_sizes, int n_in,
                              void* d_out, int out_size, void* d_ws, size_t ws_size,
                              hipStream_t stream) {
  (void)in_sizes; (void)n_in; (void)out_size; (void)ws_size;
  const float* hidden = (const float*)d_in[0];
  const float* kvs    = (const float*)d_in[1];
  const float* Wq = (const float*)d_in[2];
  const float* bq = (const float*)d_in[3];
  const float* Wk = (const float*)d_in[4];
  const float* bk = (const float*)d_in[5];
  const float* Wv = (const float*)d_in[6];
  const float* bv = (const float*)d_in[7];
  const float* Wo = (const float*)d_in[8];
  const float* bo = (const float*)d_in[9];
  float* out = (float*)d_out;

  char* ws = (char*)d_ws;
  _Float16* qbf = (_Float16*)(ws);                               //  2 MB
  _Float16* kbf = (_Float16*)(ws + (size_t)2  * 1024 * 1024);    // 16 MB
  _Float16* vtb = (_Float16*)(ws + (size_t)18 * 1024 * 1024);    // 16 MB
  float* attnOut = (float*)(ws + (size_t)34 * 1024 * 1024);      //  4 MB

  const dim3 blk(128);
  proj_kernel<0><<<dim3(E_ / 64, (B_ * T_) / 64), blk, 0, stream>>>(hidden, Wq, bq, qbf);
  proj_kernel<1><<<dim3(E_ / 64, (B_ * C_ * KV_) / 64), blk, 0, stream>>>(kvs, Wk, bk, kbf);
  proj_kernel<2><<<dim3(E_ / 64, (B_ * C_ * KV_) / 64), blk, 0, stream>>>(kvs, Wv, bv, vtb);
  attn_kernel<<<dim3(T_ / 64, H_, B_), blk, 0, stream>>>(qbf, kbf, vtb, attnOut);
  proj_out_kernel<<<dim3(E_ / 64, (B_ * T_) / 64), blk, 0, stream>>>(attnOut, Wo, bo, out);
}
